// TextLSTM_44581760532735
// MI455X (gfx1250) — matrix operand order repaired and checked by reading, not compiled, not run on hardware
//
#include <hip/hip_runtime.h>
#include <hip/hip_bf16.h>
#include <stdint.h>

#define N_CLS 50257
#define EMB_D 1024
#define HID_D 1024
#define BATCH_D 1024
#define SEQ_D 128
#define GATES_D 4096
#define NTILE_OUT 393  // ceil(50257/128)

typedef __attribute__((ext_vector_type(16))) __bf16 v16bf;
typedef __attribute__((ext_vector_type(8))) float v8f;
typedef __attribute__((ext_vector_type(8))) unsigned int v8u;

static __device__ __forceinline__ unsigned short f32_to_bf16(float f) {
  unsigned int u = __float_as_uint(f);
  unsigned int r = (u + 0x7FFFu + ((u >> 16) & 1u)) >> 16;
  return (unsigned short)r;
}

static __device__ __forceinline__ v8f zero_v8f() {
  v8f z = {0.f, 0.f, 0.f, 0.f, 0.f, 0.f, 0.f, 0.f};
  return z;
}

// two 16B reads -> one 32B (16 x bf16) WMMA fragment
static __device__ __forceinline__ v8u frag_ld(const unsigned short* p0,
                                              const unsigned short* p1) {
  uint4 q0 = *(const uint4*)p0;
  uint4 q1 = *(const uint4*)p1;
  v8u r;
  r[0] = q0.x; r[1] = q0.y; r[2] = q0.z; r[3] = q0.w;
  r[4] = q1.x; r[5] = q1.y; r[6] = q1.z; r[7] = q1.w;
  return r;
}

// async global->LDS copies (ASYNCcnt-tracked; per-lane addresses)
static __device__ __forceinline__ void async_copy_b128(
    unsigned lds_addr, const unsigned short* gsrc) {
  asm volatile("global_load_async_to_lds_b128 %0, %1, off"
               :
               : "v"(lds_addr), "v"(gsrc)
               : "memory");
}
static __device__ __forceinline__ void async_copy_b64(
    unsigned lds_addr, const unsigned short* gsrc) {
  asm volatile("global_load_async_to_lds_b64 %0, %1, off"
               :
               : "v"(lds_addr), "v"(gsrc)
               : "memory");
}
static __device__ __forceinline__ void wait_async0() {
  asm volatile("s_wait_asynccnt 0x0" ::: "memory");
}
static __device__ __forceinline__ unsigned lds_addr_of(const void* p) {
  return (unsigned)(uintptr_t)p;  // low 32 bits of the flat address give the LDS address
}

static __device__ __forceinline__ float fsigmoid(float x) {
  return 1.0f / (1.0f + __expf(-x));
}
static __device__ __forceinline__ float ftanh_fast(float x) {
  float e = __expf(2.0f * x);
  return (e - 1.0f) / (e + 1.0f);
}

// ---------------- preprocessing ----------------

// Wswz logical layout: [jb:32][kk:64][g:4][nt:2][lane:32][e:16] bf16
// B-fragment convention: lane&15 = column within 16-tile; lane<16 holds K 0..15
// of the chunk (2 per VGPR), lane>=16 holds K 16..31.
__global__ void k_swizzle_wxh(const float* __restrict__ Wx,
                              const float* __restrict__ Wh,
                              unsigned short* __restrict__ dst) {
  const long long n = 32LL * 64 * 4 * 2 * 32 * 16;  // 8,388,608
  long long i = (long long)blockIdx.x * blockDim.x + threadIdx.x;
  const long long stride = (long long)gridDim.x * blockDim.x;
  for (; i < n; i += stride) {
    const int e = (int)(i & 15);
    const int lane = (int)((i >> 4) & 31);
    const int nt = (int)((i >> 9) & 1);
    const int g = (int)((i >> 10) & 3);
    const int kk = (int)((i >> 12) & 63);
    const int jb = (int)(i >> 18);
    const int k = kk * 32 + (lane >> 4) * 16 + e;
    const int gcol = g * HID_D + jb * 32 + nt * 16 + (lane & 15);
    const float v = (k < 1024) ? Wx[(size_t)k * GATES_D + gcol]
                               : Wh[(size_t)(k - 1024) * GATES_D + gcol];
    dst[i] = f32_to_bf16(v);
  }
}

// Woswz logical layout: [jb:393][kk:32][nt:8][lane:32][e:16] bf16 (vocab tail = 0)
__global__ void k_swizzle_wout(const float* __restrict__ Wo,
                               unsigned short* __restrict__ dst) {
  const long long n = (long long)NTILE_OUT * 32 * 8 * 32 * 16;  // 51,511,296
  long long i = (long long)blockIdx.x * blockDim.x + threadIdx.x;
  const long long stride = (long long)gridDim.x * blockDim.x;
  for (; i < n; i += stride) {
    const int e = (int)(i & 15);
    const int lane = (int)((i >> 4) & 31);
    const int nt = (int)((i >> 9) & 7);
    const int kk = (int)((i >> 12) & 31);
    const int jb = (int)(i >> 17);
    const int col = jb * 128 + nt * 16 + (lane & 15);
    const int k = kk * 32 + (lane >> 4) * 16 + e;
    const float v = (col < N_CLS) ? Wo[(size_t)k * N_CLS + col] : 0.0f;
    dst[i] = f32_to_bf16(v);
  }
}

__global__ void k_bias_sum(const float* __restrict__ bx,
                           const float* __restrict__ bh,
                           float* __restrict__ bsum) {
  int i = blockIdx.x * blockDim.x + threadIdx.x;
  if (i < GATES_D) bsum[i] = bx[i] + bh[i];
}

__global__ void k_zero_u16(unsigned short* __restrict__ p, int n) {
  int i = blockIdx.x * blockDim.x + threadIdx.x;
  int stride = gridDim.x * blockDim.x;
  for (; i < n; i += stride) p[i] = 0;
}

__global__ void k_zero_f32(float* __restrict__ p, int n) {
  int i = blockIdx.x * blockDim.x + threadIdx.x;
  int stride = gridDim.x * blockDim.x;
  for (; i < n; i += stride) p[i] = 0.0f;
}

// ---------------- fused LSTM step ----------------
// Block: rows b0..b0+127, 32 gate-units across all 4 gates; K=2048
// (emb half gathered from C with packed f32->bf16, recurrent half async-copied).
// Double-buffered LDS, one barrier per K-chunk, pre-swizzled B fragments.
__global__ __launch_bounds__(256) void k_lstm_step(
    const int* __restrict__ X, const float* __restrict__ Cemb,
    const unsigned short* __restrict__ Wswz, const float* __restrict__ bias,
    const unsigned short* __restrict__ h_in, unsigned short* __restrict__ h_out,
    float* __restrict__ cst, int s) {
  __shared__ __align__(16) unsigned short sA[2][128 * 32];
  __shared__ __align__(16) unsigned short sB[2][4096];
  __shared__ int sTok[128];

  const int t = threadIdx.x;
  const int b0 = blockIdx.y * 128;
  const int jb = blockIdx.x;
  const int j0 = jb * 32;

  if (t < 128) sTok[t] = X[(b0 + t) * SEQ_D + s];

  const int lane = t & 31;
  const int lrow = lane & 15;
  const int hi = lane >> 4;
  const int m0 = (t >> 5) * 16;
  const int tr = t >> 3;
  const int tc = (t & 7) * 4;
  const unsigned short* wbase = Wswz + (size_t)jb * (64 * 4096);

  v8f acc[4][2];
#pragma unroll
  for (int g = 0; g < 4; ++g)
#pragma unroll
    for (int nt = 0; nt < 2; ++nt) acc[g][nt] = zero_v8f();

  auto stage = [&](int kchunk, int buf) {
    const int kbase = kchunk * 32;
    if (kchunk < 32) {  // embedding half: gather + packed convert (VALU path)
#pragma unroll
      for (int rr = 0; rr < 4; ++rr) {
        const int r = tr + rr * 32;
        const int tok = sTok[r];
        const float4 v = *(const float4*)&Cemb[(size_t)tok * EMB_D + kbase + tc];
        uint2 u;
        u.x = (unsigned)f32_to_bf16(v.x) | ((unsigned)f32_to_bf16(v.y) << 16);
        u.y = (unsigned)f32_to_bf16(v.z) | ((unsigned)f32_to_bf16(v.w) << 16);
        *(uint2*)&sA[buf][r * 32 + tc] = u;
      }
    } else {  // recurrent half: async global->LDS copy
      const int hk = kbase - 1024;
#pragma unroll
      for (int rr = 0; rr < 4; ++rr) {
        const int r = tr + rr * 32;
        async_copy_b64(lds_addr_of(&sA[buf][r * 32 + tc]),
                       &h_in[(size_t)(b0 + r) * HID_D + hk + tc]);
      }
    }
    // B chunk: contiguous 8KB fragment-linear async copy
    const unsigned short* bsrc = wbase + (size_t)kchunk * 4096 + t * 16;
    const unsigned ldsb = lds_addr_of(&sB[buf][t * 16]);
    async_copy_b128(ldsb, bsrc);
    async_copy_b128(ldsb + 16u, bsrc + 8);
    const int kpf = (kchunk + 2 < 64) ? (kchunk + 2) : 63;
    __builtin_prefetch(wbase + (size_t)kpf * 4096 + t * 16, 0, 0);
  };

  __syncthreads();  // sTok visible
  stage(0, 0);
  wait_async0();
  __syncthreads();

  for (int kk = 0; kk < 64; ++kk) {
    const int cur = kk & 1;
    if (kk + 1 < 64) stage(kk + 1, 1 - cur);

    // preload all fragments, then issue the WMMA chain
    const int aoff = (m0 + lrow) * 32 + (hi ? 16 : 0);
    const v8u araw = frag_ld(&sA[cur][aoff], &sA[cur][aoff + 8]);
    v8u braw[8];
#pragma unroll
    for (int i = 0; i < 8; ++i) {
      const int boff = (i * 32 + lane) * 16;
      braw[i] = frag_ld(&sB[cur][boff], &sB[cur][boff + 8]);
    }
    const v16bf afrag = __builtin_bit_cast(v16bf, araw);
#pragma unroll
    for (int g = 0; g < 4; ++g) {
#pragma unroll
      for (int nt = 0; nt < 2; ++nt) {
        const v16bf bfrag = __builtin_bit_cast(v16bf, braw[g * 2 + nt]);
        acc[g][nt] = __builtin_amdgcn_wmma_f32_16x16x32_bf16(
            false, afrag, false, bfrag, (short)0, acc[g][nt], false, false);
      }
    }
    wait_async0();  // own async copies done before barrier -> LDS globally valid
    __syncthreads();
  }

  // fused LSTM cell update (D elem r -> row m0+r+8*hi, col lrow)
#pragma unroll
  for (int nt = 0; nt < 2; ++nt) {
    const int col = j0 + nt * 16 + lrow;
    const float bi = bias[0 * HID_D + col];
    const float bf = bias[1 * HID_D + col];
    const float bg = bias[2 * HID_D + col];
    const float bo = bias[3 * HID_D + col];
#pragma unroll
    for (int r = 0; r < 8; ++r) {
      const int row = b0 + m0 + r + hi * 8;
      const float iv = fsigmoid(acc[0][nt][r] + bi);
      const float fv = fsigmoid(acc[1][nt][r] + bf);
      const float gv = ftanh_fast(acc[2][nt][r] + bg);
      const float ov = fsigmoid(acc[3][nt][r] + bo);
      const size_t idx = (size_t)row * HID_D + col;
      const float cnew = fv * cst[idx] + iv * gv;
      cst[idx] = cnew;
      h_out[idx] = f32_to_bf16(ov * ftanh_fast(cnew));
    }
  }
}

// ---------------- output projection: h @ W_out + b_out ----------------
__global__ __launch_bounds__(256) void k_logits(
    const unsigned short* __restrict__ hfin,
    const unsigned short* __restrict__ Woswz, const float* __restrict__ bout,
    float* __restrict__ out) {
  __shared__ __align__(16) unsigned short sA[2][128 * 32];
  __shared__ __align__(16) unsigned short sB[2][4096];

  const int t = threadIdx.x;
  const int b0 = blockIdx.y * 128;
  const int jb = blockIdx.x;
  const int j0 = jb * 128;

  const int lane = t & 31;
  const int lrow = lane & 15;
  const int hi = lane >> 4;
  const int m0 = (t >> 5) * 16;
  const int tr = t >> 3;
  const int tc = (t & 7) * 4;
  const unsigned short* wbase = Woswz + (size_t)jb * (32 * 4096);

  v8f acc[8];
#pragma unroll
  for (int nt = 0; nt < 8; ++nt) acc[nt] = zero_v8f();

  auto stage = [&](int kchunk, int buf) {
    const int kbase = kchunk * 32;
#pragma unroll
    for (int rr = 0; rr < 4; ++rr) {
      const int r = tr + rr * 32;
      async_copy_b64(lds_addr_of(&sA[buf][r * 32 + tc]),
                     &hfin[(size_t)(b0 + r) * HID_D + kbase + tc]);
    }
    const unsigned short* bsrc = wbase + (size_t)kchunk * 4096 + t * 16;
    const unsigned ldsb = lds_addr_of(&sB[buf][t * 16]);
    async_copy_b128(ldsb, bsrc);
    async_copy_b128(ldsb + 16u, bsrc + 8);
    const int kpf = (kchunk + 2 < 32) ? (kchunk + 2) : 31;
    __builtin_prefetch(wbase + (size_t)kpf * 4096 + t * 16, 0, 0);
  };

  stage(0, 0);
  wait_async0();
  __syncthreads();

  for (int kk = 0; kk < 32; ++kk) {
    const int cur = kk & 1;
    if (kk + 1 < 32) stage(kk + 1, 1 - cur);

    const int aoff = (m0 + lrow) * 32 + (hi ? 16 : 0);
    const v8u araw = frag_ld(&sA[cur][aoff], &sA[cur][aoff + 8]);
    v8u braw[8];
#pragma unroll
    for (int i = 0; i < 8; ++i) {
      const int boff = (i * 32 + lane) * 16;
      braw[i] = frag_ld(&sB[cur][boff], &sB[cur][boff + 8]);
    }
    const v16bf afrag = __builtin_bit_cast(v16bf, araw);
#pragma unroll
    for (int nt = 0; nt < 8; ++nt) {
      const v16bf bfrag = __builtin_bit_cast(v16bf, braw[nt]);
      acc[nt] = __builtin_amdgcn_wmma_f32_16x16x32_bf16(
          false, afrag, false, bfrag, (short)0, acc[nt], false, false);
    }
    wait_async0();
    __syncthreads();
  }

#pragma unroll
  for (int nt = 0; nt < 8; ++nt) {
    const int col = j0 + nt * 16 + lrow;
    if (col < N_CLS) {
      const float bo = bout[col];
#pragma unroll
      for (int r = 0; r < 8; ++r) {
        const int row = b0 + m0 + r + hi * 8;
        out[(size_t)row * N_CLS + col] = acc[nt][r] + bo;
      }
    }
  }
}

// ---------------- workspace layout (bytes) ----------------
#define WSWZ_OFF 0ull           // 32*64*4096*2        = 16,777,216
#define WOSWZ_OFF 16777216ull   // 393*32*4096*2       = 103,022,592
#define BIAS_OFF 119799808ull   // 4096*4
#define HA_OFF 119816192ull     // 1024*1024*2
#define HB_OFF 121913344ull     // 1024*1024*2
#define CST_OFF 124010496ull    // 1024*1024*4  (end 128,204,800)

extern "C" void kernel_launch(void* const* d_in, const int* in_sizes, int n_in,
                              void* d_out, int out_size, void* d_ws,
                              size_t ws_size, hipStream_t stream) {
  const int* X = (const int*)d_in[0];
  const float* Cemb = (const float*)d_in[1];
  const float* Wx = (const float*)d_in[2];
  const float* bx = (const float*)d_in[3];
  const float* Wh = (const float*)d_in[4];
  const float* bh = (const float*)d_in[5];
  const float* Wo = (const float*)d_in[6];
  const float* bo = (const float*)d_in[7];
  float* out = (float*)d_out;

  char* ws = (char*)d_ws;
  unsigned short* Wswz = (unsigned short*)(ws + WSWZ_OFF);
  unsigned short* Woswz = (unsigned short*)(ws + WOSWZ_OFF);
  float* bias = (float*)(ws + BIAS_OFF);
  unsigned short* hA = (unsigned short*)(ws + HA_OFF);
  unsigned short* hB = (unsigned short*)(ws + HB_OFF);
  float* cst = (float*)(ws + CST_OFF);

  // one-time (per call) weight swizzle + state init; deterministic
  k_swizzle_wxh<<<4096, 256, 0, stream>>>(Wx, Wh, Wswz);
  k_swizzle_wout<<<8192, 256, 0, stream>>>(Wo, Woswz);
  k_bias_sum<<<16, 256, 0, stream>>>(bx, bh, bias);
  k_zero_u16<<<1024, 256, 0, stream>>>(hA, HID_D * BATCH_D);
  k_zero_f32<<<1024, 256, 0, stream>>>(cst, HID_D * BATCH_D);

  // sequential recurrence: stream ordering enforces step dependencies
  for (int s = 0; s < SEQ_D; ++s) {
    const unsigned short* hin = (s & 1) ? hB : hA;
    unsigned short* hout = (s & 1) ? hA : hB;
    k_lstm_step<<<dim3(HID_D / 32, BATCH_D / 128), 256, 0, stream>>>(
        X, Cemb, Wswz, bias, hin, hout, cst, s);
  }
  // SEQ even -> final hidden state is in hA
  k_logits<<<dim3(NTILE_OUT, BATCH_D / 128), 256, 0, stream>>>(hA, Woswz, bo,
                                                               out);
}
